// Organism_79310866088138
// MI455X (gfx1250) — matrix operand order repaired and checked by reading, not compiled, not run on hardware
//
#include <hip/hip_runtime.h>

#define S_LEN 2048
#define DHID  2048
#define NRULE 8
#define KSZ   33
#define NCYC  28
#define PADK  16
#define NB    2
#define VOCAB 32000

typedef __attribute__((ext_vector_type(16))) __bf16       v16bf;
typedef __attribute__((ext_vector_type(8)))  float        v8f;
typedef __attribute__((ext_vector_type(4)))  unsigned int v4u;
typedef __attribute__((ext_vector_type(4)))  float        v4f;

union Frag16 { v4u q[2]; v16bf v; };

__device__ __forceinline__ unsigned short f2bf(float f) {
  unsigned int u = __builtin_bit_cast(unsigned int, f);
  unsigned int r = u + 0x7fffu + ((u >> 16) & 1u);   // round-to-nearest-even
  return (unsigned short)(r >> 16);
}

// async global->LDS copy (CDNA5 ASYNCcnt path; no VGPR staging)
__device__ __forceinline__ void async_b128(unsigned lds_off, const void* g) {
  asm volatile("global_load_async_to_lds_b128 %0, %1, off"
               :: "v"(lds_off), "v"((unsigned long long)(uintptr_t)g)
               : "memory");
}
__device__ __forceinline__ void wait_async0() {
  asm volatile("s_wait_asynccnt 0x0" ::: "memory");
}

// ---------------------------------------------------------------------------
// K1: per-cycle softmax -> thresholded rule weights; per-rule mix sums
// ---------------------------------------------------------------------------
__global__ void prep_kernel(const float* __restrict__ schedule,
                            const float* __restrict__ mixes,
                            float* __restrict__ weff_all,
                            float* __restrict__ mix_sums) {
  int t = threadIdx.x;
  if (t < NCYC) {
    float v[NRULE];
    float mx = -1e30f;
    for (int r = 0; r < NRULE; ++r) { v[r] = schedule[t * NRULE + r]; mx = fmaxf(mx, v[r]); }
    float sum = 0.f;
    for (int r = 0; r < NRULE; ++r) { v[r] = __expf(v[r] - mx); sum += v[r]; }
    float inv = 1.0f / sum;
    for (int r = 0; r < NRULE; ++r) {
      float w = v[r] * inv;
      weff_all[t * NRULE + r] = (w > 0.01f) ? w : 0.0f;
    }
  }
  if (t < NRULE) {
    float s = 0.f;
    for (int k = 0; k < KSZ; ++k) s += mixes[t * KSZ + k];
    mix_sums[t] = s;
  }
}

// ---------------------------------------------------------------------------
// K2: embedding gather + transpose to (B, D, S), fully coalesced both ways
// ---------------------------------------------------------------------------
__global__ void embed_transpose_kernel(const int* __restrict__ tokens,
                                       const float* __restrict__ embed_w,
                                       float* __restrict__ state) {
  __shared__ float tile[32][33];
  const int b  = blockIdx.z;
  const int d0 = blockIdx.y * 32;
  const int s0 = blockIdx.x * 32;
  const int x = threadIdx.x, y = threadIdx.y;
  #pragma unroll
  for (int i = 0; i < 4; ++i) {
    int sl = y + 8 * i;
    int tok = tokens[b * S_LEN + s0 + sl];
    tile[sl][x] = embed_w[(size_t)tok * DHID + d0 + x];   // coalesced along d
  }
  __syncthreads();
  #pragma unroll
  for (int i = 0; i < 4; ++i) {
    int dl = y + 8 * i;
    state[(size_t)(b * DHID + d0 + dl) * S_LEN + s0 + x] = tile[x][dl];  // coalesced along s
  }
}

// ---------------------------------------------------------------------------
// K3: fused 28-cycle rule scan. One block per (b,d) row; the whole scan runs
// in LDS — rows are fully independent (gm is a per-row mean).
// ---------------------------------------------------------------------------
__global__ void __launch_bounds__(256)
cycle_rows_kernel(float* __restrict__ state,
                  const float* __restrict__ kernels,
                  const float* __restrict__ weff_all,
                  const float* __restrict__ mix_sums,
                  const float* __restrict__ cycle_scale) {
  __shared__ float row[S_LEN + 2 * PADK];
  __shared__ float kern[NRULE * KSZ];
  __shared__ float weff[NCYC * NRULE];
  __shared__ float msum[NRULE];
  __shared__ float csc[NCYC];
  __shared__ float red[256];

  const int tid = threadIdx.x;
  float* grow = state + (size_t)blockIdx.x * S_LEN;

  for (int i = tid; i < NRULE * KSZ; i += 256) kern[i] = kernels[i];
  for (int i = tid; i < NCYC * NRULE; i += 256) weff[i] = weff_all[i];
  if (tid < NRULE) msum[tid] = mix_sums[tid];
  if (tid < NCYC)  csc[tid]  = cycle_scale[tid];
  if (tid < PADK) { row[tid] = 0.f; row[S_LEN + PADK + tid] = 0.f; }
  #pragma unroll
  for (int j = 0; j < S_LEN / 256; ++j) row[PADK + tid + j * 256] = grow[tid + j * 256];
  __syncthreads();

  for (int c = 0; c < NCYC; ++c) {
    // row mean
    float s = 0.f;
    #pragma unroll
    for (int j = 0; j < 8; ++j) s += row[PADK + tid + j * 256];
    red[tid] = s;
    __syncthreads();
    for (int off = 128; off > 0; off >>= 1) {
      if (tid < off) red[tid] += red[tid + off];
      __syncthreads();
    }
    const float gm = red[0] * (1.0f / (float)S_LEN);
    const float cs = csc[c];
    float gmm[NRULE], wr[NRULE];
    #pragma unroll
    for (int r = 0; r < NRULE; ++r) { gmm[r] = gm * msum[r]; wr[r] = weff[c * NRULE + r]; }

    float nv[8];
    #pragma unroll
    for (int j = 0; j < 8; ++j) {
      const int p = tid + j * 256;
      float acc[NRULE];
      #pragma unroll
      for (int r = 0; r < NRULE; ++r) acc[r] = gmm[r];
      #pragma unroll 1
      for (int k = 0; k < KSZ; ++k) {
        const float xv = row[p + k];            // halo-padded; center = PADK+p
        #pragma unroll
        for (int r = 0; r < NRULE; ++r) acc[r] = fmaf(xv, kern[r * KSZ + k], acc[r]);
      }
      float upd = 0.f;
      #pragma unroll
      for (int r = 0; r < NRULE; ++r) upd = fmaf(wr[r], tanhf(acc[r]), upd);
      nv[j] = row[PADK + p] + cs * 0.1f * upd;
    }
    __syncthreads();
    #pragma unroll
    for (int j = 0; j < 8; ++j) row[PADK + tid + j * 256] = nv[j];
    __syncthreads();
  }

  #pragma unroll
  for (int j = 0; j < 8; ++j) grow[tid + j * 256] = row[PADK + tid + j * 256];
}

// ---------------------------------------------------------------------------
// K4a: per-(b,s) mean of squares over D (coalesced along S)
// ---------------------------------------------------------------------------
__global__ void sumsq_kernel(const float* __restrict__ state, float* __restrict__ ms) {
  __shared__ float part[8][32];
  const int blk = blockIdx.x;                 // B*S/32 blocks
  const int b   = blk / (S_LEN / 32);
  const int s0  = (blk % (S_LEN / 32)) * 32;
  const int x = threadIdx.x, y = threadIdx.y;
  float acc = 0.f;
  const float* base = state + (size_t)b * DHID * S_LEN + s0 + x;
  for (int d = y; d < DHID; d += 8) {
    float v = base[(size_t)d * S_LEN];
    acc = fmaf(v, v, acc);
  }
  part[y][x] = acc;
  __syncthreads();
  if (y == 0) {
    float t = 0.f;
    #pragma unroll
    for (int yy = 0; yy < 8; ++yy) t += part[yy][x];
    ms[b * S_LEN + s0 + x] = t * (1.0f / (float)DHID);
  }
}

// ---------------------------------------------------------------------------
// K4b: RMSNorm + transpose (B,D,S)->(B*S,D) + f32->bf16
// ---------------------------------------------------------------------------
__global__ void norm_transpose_kernel(const float* __restrict__ state,
                                      const float* __restrict__ ms,
                                      const float* __restrict__ out_norm_w,
                                      unsigned short* __restrict__ Xn) {
  __shared__ float tile[32][33];
  __shared__ float rs[32];
  const int b  = blockIdx.z;
  const int d0 = blockIdx.y * 32;
  const int s0 = blockIdx.x * 32;
  const int x = threadIdx.x, y = threadIdx.y;
  if (y == 0) rs[x] = rsqrtf(ms[b * S_LEN + s0 + x] + 1.1920929e-7f);
  #pragma unroll
  for (int i = 0; i < 4; ++i) {
    int dl = y + 8 * i;
    tile[x][dl] = state[(size_t)(b * DHID + d0 + dl) * S_LEN + s0 + x];  // coalesced along s
  }
  __syncthreads();
  const float onw = out_norm_w[d0 + x];
  #pragma unroll
  for (int i = 0; i < 4; ++i) {
    int sl = y + 8 * i;
    float v = tile[sl][x] * rs[sl] * onw;
    Xn[(size_t)(b * S_LEN + s0 + sl) * DHID + d0 + x] = f2bf(v);         // coalesced along d
  }
}

// ---------------------------------------------------------------------------
// K4c: one-shot W f32 -> bf16 (makes the 131 MB weight matrix L2-resident
// for its 32x reuse across m-blocks, and turns B staging into a pure copy)
// ---------------------------------------------------------------------------
__global__ void __launch_bounds__(256)
convert_w_kernel(const float* __restrict__ W, unsigned short* __restrict__ Wb) {
  size_t i = ((size_t)blockIdx.x * 256 + threadIdx.x) * 4;
  v4f f = *(const v4f*)(W + i);
  unsigned int p0 = ((unsigned)f2bf(f.y) << 16) | f2bf(f.x);
  unsigned int p1 = ((unsigned)f2bf(f.w) << 16) | f2bf(f.z);
  unsigned int* dst = (unsigned int*)(Wb + i);
  dst[0] = p0; dst[1] = p1;
}

// ---------------------------------------------------------------------------
// K5: LM head GEMM: out(4096x32000) = Xn(4096x2048,bf16) * Wb(32000x2048)^T
// v_wmma_f32_16x16x32_bf16; 128x128 block tile, BK=32, 8 waves, 4x2 accs/wave.
// Tiles staged with global_load_async_to_lds_b128 (ASYNCcnt), double-buffered
// so the DMA for tile k+1 overlaps WMMA on tile k.
// LDS row stride = 40 bf16 (80 B): bank-conflict-free frag reads, aligned b128
// ---------------------------------------------------------------------------
#define LDT 40

__global__ void __launch_bounds__(256)
lm_head_gemm_kernel(const unsigned short* __restrict__ Xn,
                    const unsigned short* __restrict__ Wb,
                    float* __restrict__ out) {
  __shared__ unsigned short As[2][128 * LDT];
  __shared__ unsigned short Bs[2][128 * LDT];

  const int tid  = threadIdx.x;
  const int lane = tid & 31;
  const int wave = tid >> 5;
  const int wm = wave >> 2;                 // 0..1 -> 64 rows each
  const int wn = wave & 3;                  // 0..3 -> 32 cols each
  const int m0 = blockIdx.y * 128;
  const int n0 = blockIdx.x * 128;

  v8f acc[4][2] = {};

  const int lrow = lane & 15;
  const int akb  = (lane < 16) ? 0 : 8;     // A: K{0..7,16..23} | K{8..15,24..31}
  const int bkb  = (lane < 16) ? 0 : 8;     // B: K{0..7,16..23} | K{8..15,24..31}

  // issue async b128 copies for one 128x32 bf16 tile pair (4 per thread)
  auto stage = [&](int kt, int buf) {
    #pragma unroll
    for (int i = 0; i < 2; ++i) {
      int c   = tid + i * 256;              // 0..511
      int row = c >> 2;
      int col = (c & 3) * 8;
      async_b128((unsigned)(uintptr_t)&As[buf][row * LDT + col],
                 Xn + (size_t)(m0 + row) * DHID + kt + col);
      async_b128((unsigned)(uintptr_t)&Bs[buf][row * LDT + col],
                 Wb + (size_t)(n0 + row) * DHID + kt + col);
    }
  };

  stage(0, 0);
  wait_async0();
  __syncthreads();

  int cur = 0;
  for (int kt = 0; kt < DHID; kt += 32) {
    const bool more = (kt + 32) < DHID;
    if (more) stage(kt + 32, cur ^ 1);      // DMA next tile while computing

    Frag16 a[4], bf[2];
    #pragma unroll
    for (int tm = 0; tm < 4; ++tm) {
      int m = wm * 64 + tm * 16 + lrow;
      a[tm].q[0] = *(const v4u*)(&As[cur][m * LDT + akb]);
      a[tm].q[1] = *(const v4u*)(&As[cur][m * LDT + akb + 16]);
    }
    #pragma unroll
    for (int tn = 0; tn < 2; ++tn) {
      int n = wn * 32 + tn * 16 + lrow;
      bf[tn].q[0] = *(const v4u*)(&Bs[cur][n * LDT + bkb]);
      bf[tn].q[1] = *(const v4u*)(&Bs[cur][n * LDT + bkb + 16]);
    }
    #pragma unroll
    for (int tm = 0; tm < 4; ++tm)
      #pragma unroll
      for (int tn = 0; tn < 2; ++tn)
        acc[tm][tn] = __builtin_amdgcn_wmma_f32_16x16x32_bf16(
            false, a[tm].v, false, bf[tn].v, (short)0, acc[tm][tn], false, false);

    if (more) wait_async0();                // my DMAs into next buffer done
    __syncthreads();                        // everyone done reading cur buffer
    cur ^= 1;
  }

  // epilogue: C layout — lanes 0..15: N=lane, M=vgpr; lanes 16..31: M=vgpr+8
  const int nlane = lane & 15;
  const int madd  = (lane >= 16) ? 8 : 0;
  #pragma unroll
  for (int tm = 0; tm < 4; ++tm)
    #pragma unroll
    for (int tn = 0; tn < 2; ++tn) {
      int ncol = n0 + wn * 32 + tn * 16 + nlane;
      #pragma unroll
      for (int i = 0; i < 8; ++i) {
        int mrow = m0 + wm * 64 + tm * 16 + madd + i;
        out[(size_t)mrow * VOCAB + ncol] = acc[tm][tn][i];
      }
    }
}

// ---------------------------------------------------------------------------
extern "C" void kernel_launch(void* const* d_in, const int* in_sizes, int n_in,
                              void* d_out, int out_size, void* d_ws, size_t ws_size,
                              hipStream_t stream) {
  const int*   tokens      = (const int*)d_in[0];
  const float* embed_w     = (const float*)d_in[1];
  const float* kernels     = (const float*)d_in[2];
  const float* mixes       = (const float*)d_in[3];
  const float* schedule    = (const float*)d_in[4];
  const float* cycle_scale = (const float*)d_in[5];
  const float* out_norm_w  = (const float*)d_in[6];
  const float* lm_head_w   = (const float*)d_in[7];
  float* out = (float*)d_out;

  char* ws = (char*)d_ws;                       // needs ~182 MB
  size_t o = 0;
  float* state = (float*)(ws + o);              o += (size_t)NB * DHID * S_LEN * 4;   // 33.5 MB
  float* weff  = (float*)(ws + o);              o += 1024;                            // 28*8 f32
  float* msums = (float*)(ws + o);              o += 256;                             // 8 f32
  float* ms    = (float*)(ws + o);              o += (size_t)NB * S_LEN * 4;          // 16 KB
  unsigned short* Xn = (unsigned short*)(ws + o); o += (size_t)NB * S_LEN * DHID * 2; // 16.8 MB
  unsigned short* Wb = (unsigned short*)(ws + o); o += (size_t)VOCAB * DHID * 2;      // 131 MB

  prep_kernel<<<1, 32, 0, stream>>>(schedule, mixes, weff, msums);
  embed_transpose_kernel<<<dim3(S_LEN / 32, DHID / 32, NB), dim3(32, 8), 0, stream>>>(
      tokens, embed_w, state);
  cycle_rows_kernel<<<NB * DHID, 256, 0, stream>>>(state, kernels, weff, msums, cycle_scale);
  sumsq_kernel<<<NB * S_LEN / 32, dim3(32, 8), 0, stream>>>(state, ms);
  norm_transpose_kernel<<<dim3(S_LEN / 32, DHID / 32, NB), dim3(32, 8), 0, stream>>>(
      state, ms, out_norm_w, Xn);
  convert_w_kernel<<<(VOCAB * (DHID / 4)) / 256, 256, 0, stream>>>(lm_head_w, Wb);
  lm_head_gemm_kernel<<<dim3(VOCAB / 128, (NB * S_LEN) / 128), 256, 0, stream>>>(
      Xn, Wb, out);
}
